// ConceptBoxModel_28802050687820
// MI455X (gfx1250) — matrix operand order repaired and checked by reading, not compiled, not run on hardware
//
#include <hip/hip_runtime.h>
#include <hip/hip_bf16.h>
#include <math.h>

// Problem sizes (fixed by the reference)
#define BB 256   // batch
#define LL 512   // latent dim
#define NN 64    // concepts
#define DD 32    // box dim
#define CC 100   // classes
#define ND 2048  // N*D
#define ND2 4096 // N*2D  (also N*N)
#define KHEAD 8192 // 2 * ND2 fused head reduction
#define EPS 1e-6f
#define NPAD 33  // LDS row pad (gcd(33,64)=1 -> conflict-free strided access)

typedef __attribute__((ext_vector_type(16))) __bf16 v16bf;
typedef __attribute__((ext_vector_type(8)))  float  v8f;

// Convert two contiguous float4 runs (8 + 8 floats) into a v16bf A-fragment.
// A layout (16-bit, 16x32): lane m slots 0..7 = K kbase..kbase+7,
// slots 8..15 = K kbase+16..kbase+23 (kbase = 0 for lanes 0-15, 8 for 16-31).
__device__ __forceinline__ void pack_a_runs(v16bf& a, const float* run0,
                                            const float* run1) {
    const float4 a0 = *(const float4*)(run0);
    const float4 a1 = *(const float4*)(run0 + 4);
    const float4 a2 = *(const float4*)(run1);
    const float4 a3 = *(const float4*)(run1 + 4);
    a[0] = (__bf16)a0.x; a[1] = (__bf16)a0.y; a[2] = (__bf16)a0.z; a[3] = (__bf16)a0.w;
    a[4] = (__bf16)a1.x; a[5] = (__bf16)a1.y; a[6] = (__bf16)a1.z; a[7] = (__bf16)a1.w;
    a[8]  = (__bf16)a2.x; a[9]  = (__bf16)a2.y; a[10] = (__bf16)a2.z; a[11] = (__bf16)a2.w;
    a[12] = (__bf16)a3.x; a[13] = (__bf16)a3.y; a[14] = (__bf16)a3.z; a[15] = (__bf16)a3.w;
}

// ---------------------------------------------------------------------------
// Kernel 1: x_m / delta projections.
//   out[b, n*D+d] = h[b,:] . W[n,:,d] + bias[n,d]   (softplus for the offset)
// One wave per 16x16 output tile; bf16 WMMA with f32 accumulate.
// grid = (B/16, ND/16, 2), block = 32
// ---------------------------------------------------------------------------
__global__ __launch_bounds__(32) void proj_wmma_kernel(
    const float* __restrict__ h,
    const float* __restrict__ Wc, const float* __restrict__ bc,
    const float* __restrict__ Wo, const float* __restrict__ bo,
    float* __restrict__ xm, float* __restrict__ dl)
{
    const int lane   = threadIdx.x & 31;
    const int tm     = blockIdx.x;          // row tile (batch)
    const int tc     = blockIdx.y;          // col tile (n*D+d)
    const bool isOff = (blockIdx.z != 0);   // 0: centers (Wc), 1: offsets (Wo)

    const float* __restrict__ W    = isOff ? Wo : Wc;
    const float* __restrict__ bias = isOff ? bo : bc;

    const int mA     = lane & 15;
    const int kbaseA = (lane < 16) ? 0 : 8;
    const int rowA   = tm * 16 + mA;
    // B fragment: lanes 0-15 -> N=lane, K 0..7,16..23; lanes 16-31 -> N=lane-16, K 8..15,24..31.
    const int nB     = lane & 15;
    const int kbaseB = (lane < 16) ? 0 : 8;
    const int col    = tc * 16 + nB;        // flat (n*D+d) column, 0..2047

    // Weight column base: W[n, l, d] at n*L*D + l*D + d ; per-l stride = D.
    const long wbase = (long)(col >> 5) * (LL * DD) + (col & (DD - 1));
    const float* __restrict__ hrow = h + (long)rowA * LL;

    v8f acc = {};
    for (int k0 = 0; k0 < LL; k0 += 32) {
        // Keep the single-pass weight stream ahead of the WMMA loop.
        __builtin_prefetch(&W[wbase + (long)(k0 + 32 + kbaseB) * DD], 0, 1);

        v16bf a, b;
        pack_a_runs(a, hrow + k0 + kbaseA, hrow + k0 + kbaseA + 16);
#pragma unroll
        for (int s = 0; s < 16; ++s)
            b[s] = (__bf16)W[wbase + (long)(k0 + kbaseB + (s < 8 ? s : s + 8)) * DD];

        acc = __builtin_amdgcn_wmma_f32_16x16x32_bf16(false, a, false, b,
                                                      (short)0, acc, false, false);
    }

    const float bv = bias[col];
    float* __restrict__ out = isOff ? dl : xm;
    const int rbase = tm * 16 + ((lane < 16) ? 0 : 8); // C: VGPR r -> rows r / r+8
    const int ocol  = tc * 16 + (lane & 15);
#pragma unroll
    for (int r = 0; r < 8; ++r) {
        float v = acc[r] + bv;
        if (isOff) { // numerically stable softplus
            v = (v > 0.f) ? (v + log1pf(expf(-v))) : log1pf(expf(v));
        }
        out[(long)(rbase + r) * ND + ocol] = v;
    }
}

// ---------------------------------------------------------------------------
// Kernel 2: one block per batch element.
//   - stage x, delta, lo, hi in LDS (stride-33 pad -> conflict-free)
//   - p_hat[n] = sigmoid(box . Wp[n] + bp[n]),  invvol[n] = 1/prod(delta+eps)
//   - aligned[b, n*2D + {d, D+d}] = {x, delta} * p_hat[n]
//   - V[b,i,j] = prod_d(relu(min(hi_i,hi_j)-max(lo_i,lo_j))*0.5 + eps) * invvol[j]
// grid = B, block = 256
// ---------------------------------------------------------------------------
__global__ __launch_bounds__(256) void pair_kernel(
    const float* __restrict__ xm, const float* __restrict__ dl,
    const float* __restrict__ Wp, const float* __restrict__ bp,
    float* __restrict__ aligned, float* __restrict__ out_p,
    float* __restrict__ out_V)
{
    __shared__ float s_x[NN * NPAD], s_d[NN * NPAD];
    __shared__ float s_lo[NN * NPAD], s_hi[NN * NPAD];
    __shared__ float s_p[NN], s_iv[NN];

    const int b = blockIdx.x;
    const int t = threadIdx.x;

    for (int idx = t; idx < ND; idx += 256) {
        const int n = idx >> 5, d = idx & 31;
        const float x  = xm[(long)b * ND + idx];
        const float de = dl[(long)b * ND + idx];
        s_x [n * NPAD + d] = x;
        s_d [n * NPAD + d] = de;
        s_lo[n * NPAD + d] = x - de;
        s_hi[n * NPAD + d] = x + de;
    }
    __syncthreads();

    if (t < NN) {
        float s   = bp[t];
        float vol = 1.f;
#pragma unroll 8
        for (int d = 0; d < DD; ++d) {
            s  += s_x[t * NPAD + d] * Wp[t * 2 * DD + d]
                + s_d[t * NPAD + d] * Wp[t * 2 * DD + DD + d];
            vol *= (s_d[t * NPAD + d] + EPS);
        }
        const float p = 1.f / (1.f + expf(-s));
        s_p[t]  = p;
        s_iv[t] = 1.f / vol;
        out_p[b * NN + t] = p;
    }
    __syncthreads();

    for (int idx = t; idx < ND; idx += 256) {
        const int n = idx >> 5, d = idx & 31;
        const float p = s_p[n];
        aligned[(long)b * ND2 + n * 64 + d]      = s_x[n * NPAD + d] * p;
        aligned[(long)b * ND2 + n * 64 + DD + d] = s_d[n * NPAD + d] * p;
    }

    for (int pr = t; pr < NN * NN; pr += 256) {
        const int i = pr >> 6, j = pr & 63;
        float prod = 1.f;
#pragma unroll 8
        for (int d = 0; d < DD; ++d) {
            float inter = fminf(s_hi[i * NPAD + d], s_hi[j * NPAD + d])
                        - fmaxf(s_lo[i * NPAD + d], s_lo[j * NPAD + d]);
            inter = fmaxf(inter, 0.f) * 0.5f;
            prod *= (inter + EPS);
        }
        out_V[(long)b * ND2 + pr] = prod * s_iv[j];
    }
}

// ---------------------------------------------------------------------------
// Kernel 3: fused classification head.
//   y[b,c] = [aligned | V][b, 0..8191] . [W1 ; W2][:, c] + b1[c] + b2[c]
// One wave per 16x16 tile, cols padded to 112 and guarded at C=100.
// grid = (B/16, 7), block = 32
// ---------------------------------------------------------------------------
__global__ __launch_bounds__(32) void head_wmma_kernel(
    const float* __restrict__ aligned, const float* __restrict__ Vmat,
    const float* __restrict__ W1, const float* __restrict__ b1,
    const float* __restrict__ W2, const float* __restrict__ b2,
    float* __restrict__ y)
{
    const int lane   = threadIdx.x & 31;
    const int tm     = blockIdx.x;
    const int tc     = blockIdx.y;

    const int mA     = lane & 15;
    const int kbaseA = (lane < 16) ? 0 : 8;
    const int rowA   = tm * 16 + mA;
    const int nB     = lane & 15;
    const int kbaseB = (lane < 16) ? 0 : 8;
    const int col    = tc * 16 + nB;       // 0..111
    const bool colOK = (col < CC);

    const float* __restrict__ arow = aligned + (long)rowA * ND2;
    const float* __restrict__ vrow = Vmat    + (long)rowA * ND2;

    v8f acc = {};
    for (int k0 = 0; k0 < KHEAD; k0 += 32) {
        v16bf a, b;
        // Both 8-float A-runs are 8-aligned, and the aligned|V boundary (4096)
        // is 8-aligned, so each run lives wholly in one buffer.
        const int r0 = k0 + kbaseA;        // first run start
        const int r1 = r0 + 16;            // second run start
        const float* p0 = (r0 < ND2) ? (arow + r0) : (vrow + (r0 - ND2));
        const float* p1 = (r1 < ND2) ? (arow + r1) : (vrow + (r1 - ND2));
        pack_a_runs(a, p0, p1);

        const int kb0 = k0 + kbaseB;
        const float* __restrict__ wsrc =
            (kb0 < ND2) ? (W1 + (long)kb0 * CC) : (W2 + (long)(kb0 - ND2) * CC);
        __builtin_prefetch(&wsrc[32 * CC + col], 0, 1);
#pragma unroll
        for (int s = 0; s < 16; ++s) {
            float wv = 0.f;
            if (colOK) wv = wsrc[(long)(s < 8 ? s : s + 8) * CC + col];
            b[s] = (__bf16)wv;
        }
        acc = __builtin_amdgcn_wmma_f32_16x16x32_bf16(false, a, false, b,
                                                      (short)0, acc, false, false);
    }

    if (colOK) {
        const float bias = b1[col] + b2[col];
        const int rbase = tm * 16 + ((lane < 16) ? 0 : 8);
#pragma unroll
        for (int r = 0; r < 8; ++r)
            y[(long)(rbase + r) * CC + col] = acc[r] + bias;
    }
}

// ---------------------------------------------------------------------------
extern "C" void kernel_launch(void* const* d_in, const int* in_sizes, int n_in,
                              void* d_out, int out_size, void* d_ws, size_t ws_size,
                              hipStream_t stream) {
    const float* h  = (const float*)d_in[0];
    const float* Wc = (const float*)d_in[1];
    const float* bc = (const float*)d_in[2];
    const float* Wo = (const float*)d_in[3];
    const float* bo = (const float*)d_in[4];
    const float* Wp = (const float*)d_in[5];
    const float* bp = (const float*)d_in[6];
    const float* W1 = (const float*)d_in[7];
    const float* b1 = (const float*)d_in[8];
    const float* W2 = (const float*)d_in[9];
    const float* b2 = (const float*)d_in[10];

    // Workspace layout (floats): xm[B*ND] | delta[B*ND] | aligned[B*N*2D]  (8 MB)
    float* ws      = (float*)d_ws;
    float* xm      = ws;
    float* dl      = ws + (long)BB * ND;
    float* aligned = ws + (long)2 * BB * ND;

    // Output layout (floats): y_hat[B*C] | p_hat[B*N] | V[B*N*N]
    float* y     = (float*)d_out;
    float* out_p = y + (long)BB * CC;
    float* out_V = out_p + (long)BB * NN;

    proj_wmma_kernel<<<dim3(BB / 16, ND / 16, 2), 32, 0, stream>>>(
        h, Wc, bc, Wo, bo, xm, dl);

    pair_kernel<<<BB, 256, 0, stream>>>(xm, dl, Wp, bp, aligned, out_p, out_V);

    head_wmma_kernel<<<dim3(BB / 16, (CC + 15) / 16), 32, 0, stream>>>(
        aligned, out_V, W1, b1, W2, b2, y);
}
